// TulparModel_36344013259241
// MI455X (gfx1250) — compile-verified, matrix operand order checked by reading, not run on hardware
//
#include <hip/hip_runtime.h>
#include <hip/hip_bf16.h>

// ---------------------------------------------------------------------------
// Model dims
#define BB   128
#define LL   2048
#define CIN  38
#define DD   64
#define SS   32
#define PDm  32
#define FD   128
#define NH   4
#define RTOT (BB * LL)   // 262144 rows

typedef __bf16 bf16_t;
typedef __bf16 v16bf __attribute__((ext_vector_type(16)));
typedef __bf16 v8bf  __attribute__((ext_vector_type(8)));
typedef float  v8f   __attribute__((ext_vector_type(8)));

// ---------------------------------------------------------------------------
// CDNA5 async global->LDS copy (ASYNCcnt path), with safe fallback
#if defined(__gfx1250__) && __has_builtin(__builtin_amdgcn_global_load_async_to_lds_b128)
#define HAVE_ASYNC_LDS 1
#else
#define HAVE_ASYNC_LDS 0
#endif

typedef int v4i __attribute__((vector_size(16)));
typedef __attribute__((address_space(3))) v4i as3_v4i;

__device__ __forceinline__ void async_copy16(void* ldst, const void* gsrc) {
#if HAVE_ASYNC_LDS
  __builtin_amdgcn_global_load_async_to_lds_b128((v4i*)gsrc, (as3_v4i*)ldst, 0, 0);
#else
  *(uint4*)ldst = *(const uint4*)gsrc;
#endif
}
__device__ __forceinline__ void async_join() {
#if HAVE_ASYNC_LDS
#if __has_builtin(__builtin_amdgcn_s_wait_asynccnt)
  __builtin_amdgcn_s_wait_asynccnt(0);
#else
  asm volatile("s_wait_asynccnt 0x0" ::: "memory");
#endif
#endif
  __syncthreads();
}

// ---------------------------------------------------------------------------
// math helpers
__device__ __forceinline__ float gelu_exact(float x) {
  return 0.5f * x * (1.0f + erff(x * 0.70710678118654752f));
}
__device__ __forceinline__ float softplus_f(float x) {
  return (x > 20.0f) ? x : log1pf(__expf(x));
}

// ---------------------------------------------------------------------------
// WMMA helpers: fragment = two contiguous 16B runs of one bf16 LDS row
__device__ __forceinline__ v16bf frag_lds(const bf16_t* rowptr, int kg) {
  v8bf lo = *(const v8bf*)(rowptr + kg);
  v8bf hi = *(const v8bf*)(rowptr + 16 + kg);
  return __builtin_shufflevector(lo, hi, 0,1,2,3,4,5,6,7,8,9,10,11,12,13,14,15);
}
__device__ __forceinline__ v8f wmma_bf16(v16bf a, v16bf b, v8f c) {
  return __builtin_amdgcn_wmma_f32_16x16x32_bf16(false, a, false, b, (short)0, c, false, false);
}

// ---------------------------------------------------------------------------
// Kernel A: h = gelu(ln(x@Wmi+bmi)); dt = softplus(h0*Wdt+bdt); [Bm|Cm]=h@Wxp+bxp
__global__ void k_projA(const float* __restrict__ x,
                        const float* __restrict__ Wmi, const float* __restrict__ bmi,
                        const float* __restrict__ g_m, const float* __restrict__ b_m,
                        const float* __restrict__ Wdt, const float* __restrict__ bdt,
                        const float* __restrict__ Wxp, const float* __restrict__ bxp,
                        float* __restrict__ h_g, float* __restrict__ dt_g,
                        float* __restrict__ Bm_g, float* __restrict__ Cm_g) {
  __shared__ __attribute__((aligned(16))) bf16_t sx[64][72];
  __shared__ __attribute__((aligned(16))) bf16_t swm[64][72];
  __shared__ __attribute__((aligned(16))) bf16_t swx[64][72];
  __shared__ __attribute__((aligned(16))) bf16_t sh2[64][72];
  __shared__ float sraw[64][68];
  const int tid = threadIdx.x, lane = tid & 31, wv = tid >> 5;
  const int row0 = blockIdx.x * 64;
  const int kg = ((lane >> 4) & 1) << 3;
  const int mrow = lane & 15;
  const int mb = kg;

  for (int i = tid; i < 64 * 64; i += 256) {
    int r = i >> 6, c = i & 63;
    int cc = (c < CIN) ? c : (CIN - 1);
    float v = x[(size_t)(row0 + r) * CIN + cc];
    sx[r][c] = (bf16_t)((c < CIN) ? v : 0.0f);
  }
  for (int i = tid; i < 64 * 64; i += 256) {
    int n = i >> 6, k = i & 63;
    int kk = (k < CIN) ? k : (CIN - 1);
    float v = Wmi[(size_t)kk * DD + n];
    swm[n][k] = (bf16_t)((k < CIN) ? v : 0.0f);
  }
  for (int i = tid; i < 64 * 64; i += 256) {
    int n = i >> 6, k = i & 63;
    swx[n][k] = (bf16_t)Wxp[(size_t)k * 64 + n];
  }
  __syncthreads();

  // GEMM1: 4 m-tiles x 4 n-tiles = 16 wave-tiles (2 per wave, unrolled)
#pragma unroll
  for (int it = 0; it < 2; ++it) {
    const int t = wv + it * 8;
    const int mt = t >> 2, nt = t & 3;
    v8f c = {};
#pragma unroll
    for (int k0 = 0; k0 < 64; k0 += 32) {
      v16bf a = frag_lds(&sx[mt * 16 + mrow][k0], kg);
      v16bf b = frag_lds(&swm[nt * 16 + mrow][k0], kg);
      c = wmma_bf16(a, b, c);
    }
    float bb = bmi[nt * 16 + mrow];
#pragma unroll
    for (int r = 0; r < 8; ++r) sraw[mt * 16 + mb + r][nt * 16 + mrow] = c[r] + bb;
  }
  __syncthreads();

  if (tid < 64) {
    const int rr = tid;
    float mean = 0.0f;
#pragma unroll
    for (int d = 0; d < DD; ++d) mean += sraw[rr][d];
    mean *= (1.0f / DD);
    float var = 0.0f;
#pragma unroll
    for (int d = 0; d < DD; ++d) { float t = sraw[rr][d] - mean; var += t * t; }
    var *= (1.0f / DD);
    const float inv = rsqrtf(var + 1e-5f);
    const size_t gr = (size_t)(row0 + rr) * DD;
    float h0 = 0.0f;
#pragma unroll
    for (int d = 0; d < DD; ++d) {
      float v = (sraw[rr][d] - mean) * inv * g_m[d] + b_m[d];
      v = gelu_exact(v);
      sh2[rr][d] = (bf16_t)v;
      h_g[gr + d] = v;
      if (d == 0) h0 = v;
    }
#pragma unroll
    for (int d = 0; d < DD; ++d) dt_g[gr + d] = softplus_f(h0 * Wdt[d] + bdt[d]);
  }
  __syncthreads();

  // GEMM2
#pragma unroll
  for (int it = 0; it < 2; ++it) {
    const int t = wv + it * 8;
    const int mt = t >> 2, nt = t & 3;
    v8f c2 = {};
#pragma unroll
    for (int k0 = 0; k0 < 64; k0 += 32) {
      v16bf a = frag_lds(&sh2[mt * 16 + mrow][k0], kg);
      v16bf b = frag_lds(&swx[nt * 16 + mrow][k0], kg);
      c2 = wmma_bf16(a, b, c2);
    }
    const int col = nt * 16 + mrow;
    const float bb = bxp[col];
#pragma unroll
    for (int r = 0; r < 8; ++r) {
      float v = c2[r] + bb;
      size_t row = (size_t)(row0 + mt * 16 + mb + r);
      if (col < SS) Bm_g[row * SS + col] = v;
      else          Cm_g[row * SS + (col - SS)] = v;
    }
  }
}

// ---------------------------------------------------------------------------
// Kernel B: physics core -> phys[R,32] (bf16)
__device__ __forceinline__ void phys_step(const float* __restrict__ xr,
                                          const float* __restrict__ Wv, const float* __restrict__ bv,
                                          const float* __restrict__ Wr, const float* __restrict__ br,
                                          float& vmag, float& rmag, float& energy) {
  float v[3], rv[3];
#pragma unroll
  for (int o = 0; o < 3; ++o) {
    float sv = bv[o], sr = br[o];
    for (int cc = 0; cc < CIN; ++cc) {
      float xc = xr[cc];
      sv += xc * Wv[o * CIN + cc];
      sr += xc * Wr[o * CIN + cc];
    }
    v[o] = fminf(fmaxf(sv, -15.0f), 15.0f);
    float rr = softplus_f(sr);
    rv[o] = fminf(fmaxf(rr, 0.1f), 3000.0f);
  }
  vmag = sqrtf(v[0]*v[0] + v[1]*v[1] + v[2]*v[2]);
  rmag = sqrtf(rv[0]*rv[0] + rv[1]*rv[1] + rv[2]*rv[2]);
  energy = 0.5f * vmag * vmag - 1.0f / rmag;
}

__global__ void k_phys(const float* __restrict__ x,
                       const float* __restrict__ Wv, const float* __restrict__ bv,
                       const float* __restrict__ Wr, const float* __restrict__ br,
                       const float* __restrict__ Wpe, const float* __restrict__ bpe,
                       const float* __restrict__ g_pe, const float* __restrict__ be_pe,
                       bf16_t* __restrict__ phys_g) {
  const int r = blockIdx.x * blockDim.x + threadIdx.x;
  const int l = r & (LL - 1);
  const float* xr = x + (size_t)r * CIN;
  float vm, rm, en;
  phys_step(xr, Wv, bv, Wr, br, vm, rm, en);
  float ed = 0.0f;
  if (l > 0) {
    float vm0, rm0, en0;
    phys_step(xr - CIN, Wv, bv, Wr, br, vm0, rm0, en0);
    ed = en - en0;
  }
  float praw[5] = { vm, rm, en, ed, rm / vm };
  float pe[PDm];
  float mean = 0.0f;
#pragma unroll
  for (int d = 0; d < PDm; ++d) {
    float s = bpe[d];
#pragma unroll
    for (int i = 0; i < 5; ++i) s += praw[i] * Wpe[i * PDm + d];
    pe[d] = s; mean += s;
  }
  mean *= (1.0f / PDm);
  float var = 0.0f;
#pragma unroll
  for (int d = 0; d < PDm; ++d) { float t = pe[d] - mean; var += t * t; }
  var *= (1.0f / PDm);
  const float inv = rsqrtf(var + 1e-5f);
#pragma unroll
  for (int d = 0; d < PDm; ++d) {
    float v = (pe[d] - mean) * inv * g_pe[d] + be_pe[d];
    phys_g[(size_t)r * PDm + d] = (bf16_t)gelu_exact(v);
  }
}

// ---------------------------------------------------------------------------
// Kernel C: selective scan. One wave32 per (b,d); lane = state index s.
__global__ void k_scan(const float* __restrict__ A_log,
                       const float* __restrict__ dt_g, const float* __restrict__ h_g,
                       const float* __restrict__ Bm_g, const float* __restrict__ Cm_g,
                       bf16_t* __restrict__ ys_g) {
  const int lane = threadIdx.x & 31;
  const int wid = blockIdx.x * (blockDim.x >> 5) + (threadIdx.x >> 5);
  const int b = wid >> 6;
  const int d = wid & (DD - 1);
  const float Ac = -__expf(A_log[d * SS + lane]);
  float state = 0.0f;
  const size_t base = (size_t)b * LL;
  for (int l = 0; l < LL; ++l) {
    const size_t r = base + l;
    __builtin_prefetch(dt_g + (r + 32) * DD + d, 0, 1);
    __builtin_prefetch(h_g  + (r + 32) * DD + d, 0, 1);
    __builtin_prefetch(Bm_g + (r + 32) * SS + lane, 0, 1);
    __builtin_prefetch(Cm_g + (r + 32) * SS + lane, 0, 1);
    const float dtv = dt_g[r * DD + d];
    const float hv  = h_g[r * DD + d];
    const float Bv  = Bm_g[r * SS + lane];
    const float Cv  = Cm_g[r * SS + lane];
    const float dA  = __expf(Ac * dtv);
    state = dA * state + dtv * hv * Bv;
    float y = state * Cv;
#pragma unroll
    for (int off = 16; off > 0; off >>= 1) y += __shfl_xor(y, off, 32);
    if (lane == 0) ys_g[r * DD + d] = (bf16_t)y;
  }
}

// ---------------------------------------------------------------------------
// Kernel D: mamba = ys@Wmo+bmo; kv = concat(phys, mamba)@Wamp + bamp
__global__ void k_fuse(const bf16_t* __restrict__ ys_g,
                       const float* __restrict__ Wmo, const float* __restrict__ bmo,
                       const bf16_t* __restrict__ phys_g,
                       const float* __restrict__ Wamp, const float* __restrict__ bamp,
                       bf16_t* __restrict__ kv_g) {
  __shared__ __attribute__((aligned(16))) bf16_t sys[64][72];
  __shared__ __attribute__((aligned(16))) bf16_t swmo[64][72];
  __shared__ __attribute__((aligned(16))) bf16_t scat[64][104];
  __shared__ __attribute__((aligned(16))) bf16_t swamp[128][104];
  const int tid = threadIdx.x, lane = tid & 31, wv = tid >> 5;
  const int row0 = blockIdx.x * 64;
  const int kg = ((lane >> 4) & 1) << 3;
  const int mrow = lane & 15;
  const int mb = kg;

  // async stage ys tile (8x16B per row) and phys tile (4x16B per row)
  for (int i = tid; i < 64 * 8; i += 256) {
    int r = i >> 3, q = i & 7;
    async_copy16(&sys[r][q * 8], ys_g + (size_t)(row0 + r) * DD + q * 8);
  }
  for (int i = tid; i < 64 * 4; i += 256) {
    int r = i >> 2, q = i & 3;
    async_copy16(&scat[r][q * 8], phys_g + (size_t)(row0 + r) * PDm + q * 8);
  }
  // stage Wmo^T, Wamp^T (transpose + cvt, regular path)
  for (int i = tid; i < 64 * 64; i += 256) {
    int n = i >> 6, k = i & 63;
    swmo[n][k] = (bf16_t)Wmo[(size_t)k * DD + n];
  }
  for (int i = tid; i < 128 * 96; i += 256) {
    int n = i / 96, k = i % 96;
    swamp[n][k] = (bf16_t)Wamp[(size_t)k * FD + n];
  }
  async_join();

  // GEMM1: mamba -> scat cols 32..95 ; 4mt x 4nt (2 tiles per wave)
#pragma unroll
  for (int it = 0; it < 2; ++it) {
    const int t = wv + it * 8;
    const int mt = t >> 2, nt = t & 3;
    v8f c = {};
#pragma unroll
    for (int k0 = 0; k0 < 64; k0 += 32) {
      v16bf a = frag_lds(&sys[mt * 16 + mrow][k0], kg);
      v16bf b = frag_lds(&swmo[nt * 16 + mrow][k0], kg);
      c = wmma_bf16(a, b, c);
    }
    float bb = bmo[nt * 16 + mrow];
#pragma unroll
    for (int r = 0; r < 8; ++r)
      scat[mt * 16 + mb + r][32 + nt * 16 + mrow] = (bf16_t)(c[r] + bb);
  }
  __syncthreads();

  // GEMM2: kv = scat[64x96] @ Wamp[96x128] ; 4mt x 8nt (4 tiles per wave)
#pragma unroll
  for (int it = 0; it < 4; ++it) {
    const int t = wv + it * 8;
    const int mt = t >> 3, nt = t & 7;
    v8f c2 = {};
#pragma unroll
    for (int k0 = 0; k0 < 96; k0 += 32) {
      v16bf a = frag_lds(&scat[mt * 16 + mrow][k0], kg);
      v16bf b = frag_lds(&swamp[nt * 16 + mrow][k0], kg);
      c2 = wmma_bf16(a, b, c2);
    }
    const float bb = bamp[nt * 16 + mrow];
#pragma unroll
    for (int r = 0; r < 8; ++r)
      kv_g[(size_t)(row0 + mt * 16 + mb + r) * FD + nt * 16 + mrow] = (bf16_t)(c2[r] + bb);
  }
}

// ---------------------------------------------------------------------------
// Kernel E: per-batch query chain: mean(x) -> Wch -> Wac -> Wq  => qh[B,128]
__global__ void k_query(const float* __restrict__ x,
                        const float* __restrict__ Wch, const float* __restrict__ bch,
                        const float* __restrict__ Wac, const float* __restrict__ bac,
                        const float* __restrict__ Wq,  const float* __restrict__ bq,
                        float* __restrict__ qh_g) {
  __shared__ float smean[CIN];
  __shared__ float sfeat[64];
  __shared__ float sq[FD];
  const int b = blockIdx.x, t = threadIdx.x;
  if (t < CIN) {
    float s = 0.0f;
    const float* xb = x + (size_t)b * LL * CIN + t;
    for (int l = 0; l < LL; ++l) s += xb[(size_t)l * CIN];
    smean[t] = s * (1.0f / LL);
  }
  __syncthreads();
  if (t < 64) {
    float s = bch[t];
    for (int cc = 0; cc < CIN; ++cc) s += smean[cc] * Wch[cc * 64 + t];
    sfeat[t] = s;
  }
  __syncthreads();
  if (t < FD) {
    float s = bac[t];
    for (int cc = 0; cc < 64; ++cc) s += sfeat[cc] * Wac[cc * FD + t];
    sq[t] = s;
  }
  __syncthreads();
  if (t < FD) {
    float s = bq[t];
    for (int cc = 0; cc < FD; ++cc) s += sq[cc] * Wq[cc * FD + t];
    qh_g[b * FD + t] = s;
  }
}

// ---------------------------------------------------------------------------
// Kernel F1: kh = kv@Wk+bk, vh = kv@Wva+bva ; tile = 64 rows
__global__ void k_kvproj(const bf16_t* __restrict__ kv_g,
                         const float* __restrict__ Wk,  const float* __restrict__ bk,
                         const float* __restrict__ Wva, const float* __restrict__ bva,
                         bf16_t* __restrict__ kh_g, bf16_t* __restrict__ vh_g) {
  __shared__ __attribute__((aligned(16))) bf16_t skv[64][136];
  __shared__ __attribute__((aligned(16))) bf16_t swk[128][136];
  __shared__ __attribute__((aligned(16))) bf16_t swv[128][136];
  const int tid = threadIdx.x, lane = tid & 31, wv = tid >> 5;
  const int row0 = blockIdx.x * 64;
  const int kg = ((lane >> 4) & 1) << 3;
  const int mrow = lane & 15;
  const int mb = kg;

  // async stage kv tile (16x16B per row)
  for (int i = tid; i < 64 * 16; i += 256) {
    int r = i >> 4, q = i & 15;
    async_copy16(&skv[r][q * 8], kv_g + (size_t)(row0 + r) * FD + q * 8);
  }
  for (int i = tid; i < 128 * 128; i += 256) {
    int n = i >> 7, k = i & 127;
    swk[n][k] = (bf16_t)Wk[(size_t)k * FD + n];
    swv[n][k] = (bf16_t)Wva[(size_t)k * FD + n];
  }
  async_join();

  // 4 m-tiles x 8 n-tiles = 32 wave-tiles (4 per wave); K and V share A frags
#pragma unroll
  for (int it = 0; it < 4; ++it) {
    const int t = wv + it * 8;
    const int mt = t >> 3, nt = t & 7;
    v8f ck = {}, cv = {};
#pragma unroll
    for (int k0 = 0; k0 < FD; k0 += 32) {
      v16bf a  = frag_lds(&skv[mt * 16 + mrow][k0], kg);
      v16bf b1 = frag_lds(&swk[nt * 16 + mrow][k0], kg);
      ck = wmma_bf16(a, b1, ck);
      v16bf b2 = frag_lds(&swv[nt * 16 + mrow][k0], kg);
      cv = wmma_bf16(a, b2, cv);
    }
    const float bbk = bk[nt * 16 + mrow], bbv = bva[nt * 16 + mrow];
#pragma unroll
    for (int r = 0; r < 8; ++r) {
      size_t ro = (size_t)(row0 + mt * 16 + mb + r) * FD + nt * 16 + mrow;
      kh_g[ro] = (bf16_t)(ck[r] + bbk);
      vh_g[ro] = (bf16_t)(cv[r] + bbv);
    }
  }
}

// ---------------------------------------------------------------------------
// Kernel F2: single-query online-softmax attention.
// One block (4 waves) per (b,head); each wave scans L/4, partials merged in LDS.
__global__ void k_attn(const float* __restrict__ qh_g,
                       const bf16_t* __restrict__ kh_g, const bf16_t* __restrict__ vh_g,
                       float* __restrict__ o_g) {
  __shared__ float sm[4], sl[4], sacc[4][32];
  const int tid = threadIdx.x, lane = tid & 31, w = tid >> 5;
  const int b = blockIdx.x >> 2, hh = blockIdx.x & 3;
  const float q = qh_g[b * FD + hh * 32 + lane];
  float m = -1e30f, lsum = 0.0f, acc = 0.0f;
  const size_t base = (size_t)b * LL;
  const int l0 = w * (LL / 4), l1 = l0 + (LL / 4);
  for (int l = l0; l < l1; ++l) {
    const size_t ro = (base + l) * FD + hh * 32 + lane;
    __builtin_prefetch(kh_g + ro + 16 * FD, 0, 1);
    __builtin_prefetch(vh_g + ro + 16 * FD, 0, 1);
    float s = q * (float)kh_g[ro];
#pragma unroll
    for (int off = 16; off > 0; off >>= 1) s += __shfl_xor(s, off, 32);
    s *= 0.17677669529663689f;     // 1/sqrt(32)
    const float nm = fmaxf(m, s);
    const float corr = __expf(m - nm);
    const float p = __expf(s - nm);
    const float vd = (float)vh_g[ro];
    lsum = lsum * corr + p;
    acc  = acc * corr + p * vd;
    m = nm;
  }
  if (lane == 0) { sm[w] = m; sl[w] = lsum; }
  sacc[w][lane] = acc;
  __syncthreads();
  if (w == 0) {
    float M = fmaxf(fmaxf(sm[0], sm[1]), fmaxf(sm[2], sm[3]));
    float ltot = 0.0f, atot = 0.0f;
#pragma unroll
    for (int i = 0; i < 4; ++i) {
      float e = __expf(sm[i] - M);
      ltot += sl[i] * e;
      atot += sacc[i][lane] * e;
    }
    o_g[b * FD + hh * 32 + lane] = atot / ltot;
  }
}

// ---------------------------------------------------------------------------
// Kernel G: o@Wo+bo -> gelu(@Wc1+bc1) -> @Wc2+bc2 ; one block (128 thr) per b
__global__ void k_head(const float* __restrict__ o_g,
                       const float* __restrict__ Wo,  const float* __restrict__ bo,
                       const float* __restrict__ Wc1, const float* __restrict__ bc1,
                       const float* __restrict__ Wc2, const float* __restrict__ bc2,
                       float* __restrict__ out) {
  __shared__ float s1[FD], s2[FD], s3[64];
  const int b = blockIdx.x, t = threadIdx.x;
  s1[t] = o_g[b * FD + t];
  __syncthreads();
  {
    float s = bo[t];
    for (int cc = 0; cc < FD; ++cc) s += s1[cc] * Wo[cc * FD + t];
    s2[t] = s;
  }
  __syncthreads();
  if (t < 64) {
    float v = bc1[t];
    for (int cc = 0; cc < FD; ++cc) v += s2[cc] * Wc1[cc * 64 + t];
    s3[t] = gelu_exact(v);
  }
  __syncthreads();
  if (t == 0) {
    float v = bc2[0];
    for (int cc = 0; cc < 64; ++cc) v += s3[cc] * Wc2[cc];
    out[b] = v;
  }
}

// ---------------------------------------------------------------------------
extern "C" void kernel_launch(void* const* d_in, const int* in_sizes, int n_in,
                              void* d_out, int out_size, void* d_ws, size_t ws_size,
                              hipStream_t stream) {
  const float* x    = (const float*)d_in[0];
  const float* Wv   = (const float*)d_in[1];
  const float* bv   = (const float*)d_in[2];
  const float* Wr   = (const float*)d_in[3];
  const float* br   = (const float*)d_in[4];
  const float* Wpe  = (const float*)d_in[5];
  const float* bpe  = (const float*)d_in[6];
  const float* g_pe = (const float*)d_in[7];
  const float* be_pe= (const float*)d_in[8];
  const float* Wmi  = (const float*)d_in[9];
  const float* bmi  = (const float*)d_in[10];
  const float* g_m  = (const float*)d_in[11];
  const float* b_m  = (const float*)d_in[12];
  const float* A_log= (const float*)d_in[13];
  const float* Wdt  = (const float*)d_in[14];
  const float* bdt  = (const float*)d_in[15];
  const float* Wxp  = (const float*)d_in[16];
  const float* bxp  = (const float*)d_in[17];
  const float* Wmo  = (const float*)d_in[18];
  const float* bmo  = (const float*)d_in[19];
  const float* Wch  = (const float*)d_in[20];
  const float* bch  = (const float*)d_in[21];
  const float* Wac  = (const float*)d_in[22];
  const float* bac  = (const float*)d_in[23];
  const float* Wamp = (const float*)d_in[24];
  const float* bamp = (const float*)d_in[25];
  const float* Wq   = (const float*)d_in[26];
  const float* bq   = (const float*)d_in[27];
  const float* Wk   = (const float*)d_in[28];
  const float* bk   = (const float*)d_in[29];
  const float* Wva  = (const float*)d_in[30];
  const float* bva  = (const float*)d_in[31];
  const float* Wo   = (const float*)d_in[32];
  const float* bo   = (const float*)d_in[33];
  const float* Wc1  = (const float*)d_in[34];
  const float* bc1  = (const float*)d_in[35];
  const float* Wc2  = (const float*)d_in[36];
  const float* bc2  = (const float*)d_in[37];

  // workspace layout (bytes); kh/vh alias h/dt regions freed after the scan
  char* ws = (char*)d_ws;
  float*  h_g    = (float*)(ws + 0);                         // 64 MB
  float*  dt_g   = (float*)(ws + 67108864ull);               // 64 MB
  float*  Bm_g   = (float*)(ws + 134217728ull);              // 32 MB
  float*  Cm_g   = (float*)(ws + 167772160ull);              // 32 MB
  bf16_t* phys_g = (bf16_t*)(ws + 201326592ull);             // 16 MB
  bf16_t* ys_g   = (bf16_t*)(ws + 218103808ull);             // 32 MB
  bf16_t* kv_g   = (bf16_t*)(ws + 251658240ull);             // 64 MB
  float*  qh_g   = (float*)(ws + 318767104ull);              // 64 KB
  float*  o_g    = (float*)(ws + 318832640ull);              // 64 KB
  bf16_t* kh_g   = (bf16_t*)(ws + 0);                        // alias h
  bf16_t* vh_g   = (bf16_t*)(ws + 67108864ull);              // alias dt

  float* out = (float*)d_out;

  k_projA<<<RTOT / 64, 256, 0, stream>>>(x, Wmi, bmi, g_m, b_m, Wdt, bdt, Wxp, bxp,
                                         h_g, dt_g, Bm_g, Cm_g);
  k_phys<<<RTOT / 256, 256, 0, stream>>>(x, Wv, bv, Wr, br, Wpe, bpe, g_pe, be_pe, phys_g);
  k_scan<<<(BB * DD) / 8, 256, 0, stream>>>(A_log, dt_g, h_g, Bm_g, Cm_g, ys_g);
  k_fuse<<<RTOT / 64, 256, 0, stream>>>(ys_g, Wmo, bmo, phys_g, Wamp, bamp, kv_g);
  k_query<<<BB, 256, 0, stream>>>(x, Wch, bch, Wac, bac, Wq, bq, qh_g);
  k_kvproj<<<RTOT / 64, 256, 0, stream>>>(kv_g, Wk, bk, Wva, bva, kh_g, vh_g);
  k_attn<<<BB * NH, 128, 0, stream>>>(qh_g, kh_g, vh_g, o_g);
  k_head<<<BB, 128, 0, stream>>>(o_g, Wo, bo, Wc1, bc1, Wc2, bc2, out);
}
